// MixingNetwork_1623497638282
// MI455X (gfx1250) — matrix operand order repaired and checked by reading, not compiled, not run on hardware
//
#include <hip/hip_runtime.h>
#include <cmath>

#define NEDGE   131072
#define NNODE   8192
#define NGRP    1024

// ---------- types / helpers ----------
typedef __attribute__((ext_vector_type(16))) __bf16 v16bf;
typedef __attribute__((ext_vector_type(8)))  float  v8f;

__device__ __forceinline__ unsigned short f2bf(float f) {
  unsigned u = __builtin_bit_cast(unsigned, f);
  unsigned r = (u + 0x7FFFu + ((u >> 16) & 1u)) >> 16;
  return (unsigned short)r;
}
__device__ __forceinline__ float bf2f(unsigned short h) {
  unsigned u = ((unsigned)h) << 16;
  return __builtin_bit_cast(float, u);
}
__device__ __forceinline__ __bf16 bfbits(unsigned short h) {
  return __builtin_bit_cast(__bf16, h);
}
__device__ __forceinline__ float silu_f(float x) { return x / (1.0f + __expf(-x)); }
__device__ __forceinline__ float sigm_f(float x) { return 1.0f / (1.0f + __expf(-x)); }

// ---------- edge embedding: sh (sqrt3*unit) + cosine radial basis (bf16, K padded to 32) ----------
__global__ void edge_embed_kernel(const float* __restrict__ pos, const float* __restrict__ shift,
                                  const float* __restrict__ lat, const int* __restrict__ batch,
                                  const int* __restrict__ esrc, const int* __restrict__ edst,
                                  float* __restrict__ sh1o, unsigned short* __restrict__ basis, int nE) {
  int e = blockIdx.x * blockDim.x + threadIdx.x;
  if (e >= nE) return;
  int s = esrc[e], t = edst[e];
  const float* L = lat + (size_t)batch[s] * 9;
  float s0 = shift[e*3+0], s1 = shift[e*3+1], s2 = shift[e*3+2];
  float ev[3];
#pragma unroll
  for (int j = 0; j < 3; ++j)
    ev[j] = pos[t*3+j] - pos[s*3+j] + s0*L[j] + s1*L[3+j] + s2*L[6+j];
  float r = sqrtf(ev[0]*ev[0] + ev[1]*ev[1] + ev[2]*ev[2] + 1e-12f);
  float inv = 1.0f / r;
#pragma unroll
  for (int j = 0; j < 3; ++j) sh1o[e*3+j] = 1.7320508075688772f * ev[j] * inv;
  unsigned short* bp = basis + (size_t)e * 32;
  const float istep = 17.0f / 5.0f;   // 1/step, values_j = 5*(j+1)/17
#pragma unroll
  for (int j = 0; j < 16; ++j) {
    float diff = r * istep - (float)(j + 1);
    float v = (diff > -1.0f && diff < 1.0f) ? cosf(1.5707963267948966f * diff) * 4.0f : 0.0f;
    bp[j] = f2bf(v);
  }
#pragma unroll
  for (int j = 16; j < 32; ++j) bp[j] = 0;
}

// ---------- pack B (f32, rows x N, row-major) into WMMA-fragment-swizzled bf16 ----------
// dst index = ((ntile*(Kp/32) + kb)*32 + lane)*16 + e ; lane: n = ntile*16+(lane&15),
// k = kb*32 + (lane>>4)*16 + e.  Each lane's 16 values are contiguous -> 2x b128 loads in GEMM.
__global__ void pack_b_bf16(const float* __restrict__ src, unsigned short* __restrict__ dst,
                            int rows, int N, int Kp, float scale, int total) {
  int tid = blockIdx.x * blockDim.x + threadIdx.x;
  if (tid >= total) return;
  int e    = tid & 15;
  int lane = (tid >> 4) & 31;
  int blk  = tid >> 9;
  int kblocks = Kp >> 5;
  int kb    = blk % kblocks;
  int ntile = blk / kblocks;
  int k = kb * 32 + (lane >> 4) * 16 + e;
  int n = ntile * 16 + (lane & 15);
  float v = (k < rows) ? src[(size_t)k * N + n] * scale : 0.0f;
  dst[tid] = f2bf(v);
}

// ---------- WMMA bf16 GEMM: C(MxN) = A(MxK,bf16 row-major) * B(swizzled bf16) ----------
// 256 threads = 8 waves; wave w -> M tile (blockIdx.x*8+w), N tile = blockIdx.y.
// A frag: lane m=l&15, half=l>>4; e 0..15 -> k=k0+half*16+e
//   -> 32 contiguous bytes per lane (vectorizes to b128).
// B frag: lane n=l&15, half=l>>4; e -> k = k0+half*16+e -> 32 contiguous bytes in swizzled buffer.
// C/D:    vgpr j: m = j + 8*half, n = l&15
__global__ __launch_bounds__(256)
void gemm_bf16_wmma_kernel(const unsigned short* __restrict__ A, int lda,
                           const unsigned short* __restrict__ Bs,
                           void* __restrict__ Cout, int ldc,
                           int K, int act_silu, int store_bf16) {
  const int lane  = threadIdx.x & 31;
  const int wave  = threadIdx.x >> 5;
  const int tileM = blockIdx.x * 8 + wave;
  const int tileN = blockIdx.y;
  const int half  = lane >> 4, l15 = lane & 15;
  const int kblocks = K >> 5;
  const unsigned short* Ap = A + ((size_t)tileM * 16 + l15) * lda + half * 16;
  const unsigned short* Bp = Bs + (((size_t)tileN * kblocks) * 32 + lane) * 16;
  v8f acc = {0.f, 0.f, 0.f, 0.f, 0.f, 0.f, 0.f, 0.f};
  for (int kb = 0; kb < kblocks; ++kb) {
    v16bf a, b;
#pragma unroll
    for (int e = 0; e < 8; ++e) {
      a[e]     = bfbits(Ap[e]);
      a[e + 8] = bfbits(Ap[8 + e]);
    }
#pragma unroll
    for (int e = 0; e < 16; ++e) b[e] = bfbits(Bp[e]);
    acc = __builtin_amdgcn_wmma_f32_16x16x32_bf16(false, a, false, b, (short)0, acc, false, false);
    Ap += 32;
    Bp += 512;
  }
#pragma unroll
  for (int j = 0; j < 8; ++j) {
    size_t m = (size_t)tileM * 16 + j + half * 8;
    int n = tileN * 16 + l15;
    float v = acc[j];
    if (act_silu) v = silu_f(v);
    if (store_bf16) ((unsigned short*)Cout)[m * ldc + n] = f2bf(v);
    else            ((float*)Cout)[m * ldc + n] = v;
  }
}

// ---------- node linear: out[n,o,i] = scale * sum_u in[n,u,i]*W[o,u]  (+ optional cs*sc + cx*lin) ----------
__global__ void lin_node_kernel(const float* __restrict__ in, int istride, int ioff,
                                const float* __restrict__ W, int O, int U, int dim, float scale,
                                float* __restrict__ out, int ostride, int ooff,
                                const float* __restrict__ sc, int scstride, int scoff,
                                float cs, float cx, int total) {
  int tid = blockIdx.x * blockDim.x + threadIdx.x;
  if (tid >= total) return;
  int i = tid % dim; int rem = tid / dim;
  int o = rem % O;   int n = rem / O;
  const float* ip = in + (size_t)n * istride + ioff + i;
  const float* wp = W + (size_t)o * U;
  float acc = 0.0f;
  for (int u = 0; u < U; ++u) acc += ip[u * dim] * wp[u];
  float v = acc * scale;
  if (sc) v = cs * sc[(size_t)n * scstride + scoff + o * dim + i] + cx * v;
  out[(size_t)n * ostride + ooff + o * dim + i] = v;
}

// ---------- tensor product + scatter (segment sum via f32 atomics) ----------
struct TPIns { int l1, l2, l3, mul, foff, woff, aoff; };
struct TPDesc { int n, fstride, astride, wld, totmul; TPIns ins[8]; };

__global__ void tp_scatter_kernel(const int* __restrict__ esrc, const int* __restrict__ edst,
                                  const float* __restrict__ sh1o, const unsigned short* __restrict__ wbuf,
                                  const float* __restrict__ feat, float* __restrict__ agg,
                                  TPDesc d, int total) {
  int tid = blockIdx.x * blockDim.x + threadIdx.x;
  if (tid >= total) return;
  int c = tid % d.totmul;
  int e = tid / d.totmul;
  int k = 0;
  while (k < d.n - 1 && c >= d.ins[k].mul) { c -= d.ins[k].mul; ++k; }
  const TPIns I = d.ins[k];
  const int s = esrc[e], t = edst[e];
  float w = bf2f(wbuf[(size_t)e * d.wld + I.woff + c]) * 0.25f;  // 1/sqrt(16 neighbors)
  const float* xp = feat + (size_t)s * d.fstride + I.foff + c * (I.l1 ? 3 : 1);
  float* ap = agg + (size_t)t * d.astride + I.aoff + c * (I.l3 ? 3 : 1);
  if (!I.l1 && !I.l2) {                        // 0e x 0e -> 0e
    atomicAdd(ap, w * xp[0]);
  } else if (!I.l1) {                          // 0e x 1o -> 1o
    float x0 = xp[0];
#pragma unroll
    for (int i = 0; i < 3; ++i) atomicAdd(ap + i, w * x0 * sh1o[e*3+i]);
  } else if (!I.l2) {                          // 1x x 0e -> 1x   (sh0e == 1)
#pragma unroll
    for (int i = 0; i < 3; ++i) atomicAdd(ap + i, w * xp[i]);
  } else if (!I.l3) {                          // 1x x 1o -> 0x : dot/sqrt3
    float dv = (xp[0]*sh1o[e*3+0] + xp[1]*sh1o[e*3+1] + xp[2]*sh1o[e*3+2]) * 0.57735026918962576f;
    atomicAdd(ap, w * dv);
  } else {                                     // 1x x 1o -> 1x : cross/sqrt2
    float y0 = sh1o[e*3+0], y1 = sh1o[e*3+1], y2 = sh1o[e*3+2];
    atomicAdd(ap + 0, w * (xp[1]*y2 - xp[2]*y1) * 0.70710678118654752f);
    atomicAdd(ap + 1, w * (xp[2]*y0 - xp[0]*y2) * 0.70710678118654752f);
    atomicAdd(ap + 2, w * (xp[0]*y1 - xp[1]*y0) * 0.70710678118654752f);
  }
}

// ---------- gates ----------
__global__ void gate1_kernel(const float* __restrict__ Y, float* __restrict__ G, int total) {
  int tid = blockIdx.x * blockDim.x + threadIdx.x;
  if (tid >= total) return;
  int c = tid & 31, n = tid >> 5;
  const float* y = Y + (size_t)n * 288;   // 0e@0(96), 1o@96, 1e@192
  float* g = G + (size_t)n * 224;         // 0e@0(32), 1o@32, 1e@128
  g[c] = silu_f(y[c]);
  float a = sigm_f(y[32 + c]), b = sigm_f(y[64 + c]);
#pragma unroll
  for (int i = 0; i < 3; ++i) {
    g[32  + c*3 + i] = y[96  + c*3 + i] * a;
    g[128 + c*3 + i] = y[192 + c*3 + i] * b;
  }
}
__global__ void gate2_kernel(const float* __restrict__ Y, float* __restrict__ G, int total) {
  int tid = blockIdx.x * blockDim.x + threadIdx.x;
  if (tid >= total) return;
  int c = tid & 31, n = tid >> 5;
  const float* y = Y + (size_t)n * 320;   // 0o@0(32), 0e@32(96), 1o@128, 1e@224
  float* g = G + (size_t)n * 256;         // 0o@0, 0e@32, 1o@64, 1e@160
  g[c]      = tanhf(y[c]);
  g[32 + c] = silu_f(y[32 + c]);
  float a = sigm_f(y[64 + c]), b = sigm_f(y[96 + c]);
#pragma unroll
  for (int i = 0; i < 3; ++i) {
    g[64  + c*3 + i] = y[128 + c*3 + i] * a;
    g[160 + c*3 + i] = y[224 + c*3 + i] * b;
  }
}

// ---------- mix (group mean concat) ----------
__global__ void count_kernel(const int* __restrict__ gidx, float* __restrict__ cnt, int n) {
  int tid = blockIdx.x * blockDim.x + threadIdx.x;
  if (tid < n) atomicAdd(&cnt[gidx[tid]], 1.0f);
}
__global__ void gsum_scatter_kernel(const float* __restrict__ gated, const int* __restrict__ gidx,
                                    float* __restrict__ gsum, int gstride, int total) {
  int tid = blockIdx.x * blockDim.x + threadIdx.x;
  if (tid >= total) return;
  int n = tid / gstride, j = tid % gstride;
  atomicAdd(&gsum[(size_t)gidx[n] * gstride + j], gated[tid]);
}
__global__ void mix_fill_kernel(const float* __restrict__ gated, int gstride, int goff,
                                const float* __restrict__ gsum, const float* __restrict__ cnt,
                                const int* __restrict__ gidx,
                                float* __restrict__ feats, int fstride, int foff,
                                int ch, int dim, int total) {
  int tid = blockIdx.x * blockDim.x + threadIdx.x;
  if (tid >= total) return;
  int i = tid % dim; int rem = tid / dim;
  int c = rem % ch;  int n = rem / ch;
  int g = gidx[n];
  float cc = cnt[g]; if (cc < 1.0f) cc = 1.0f;
  float v = gated[(size_t)n * gstride + goff + c*dim + i];
  float m = gsum[(size_t)g * gstride + goff + c*dim + i] / cc;
  feats[(size_t)n * fstride + foff + c*dim + i]        = v;
  feats[(size_t)n * fstride + foff + (ch + c)*dim + i] = m;
}

// ---------- host helpers ----------
static inline void launch_lin(hipStream_t st, const float* in, int istride, int ioff,
                              const float* W, int O, int U, int dim,
                              float* out, int ostride, int ooff,
                              const float* sc, int scstride, int scoff, float cs, float cx) {
  int total = NNODE * O * dim;
  lin_node_kernel<<<(total + 255) / 256, 256, 0, st>>>(in, istride, ioff, W, O, U, dim,
      1.0f / sqrtf((float)U), out, ostride, ooff, sc, scstride, scoff, cs, cx, total);
}
static inline void launch_mix(hipStream_t st, const float* gated, int gstride, int goff,
                              const float* gsum, const float* cnt, const int* gidx,
                              float* feats, int fstride, int foff, int ch, int dim) {
  int total = NNODE * ch * dim;
  mix_fill_kernel<<<(total + 255) / 256, 256, 0, st>>>(gated, gstride, goff, gsum, cnt, gidx,
      feats, fstride, foff, ch, dim, total);
}
static inline void launch_packb(hipStream_t st, const float* src, unsigned short* dst,
                                int rows, int N, int Kp, float scale) {
  int total = N * Kp;  // (N/16)*(Kp/32)*512
  pack_b_bf16<<<(total + 255) / 256, 256, 0, st>>>(src, dst, rows, N, Kp, scale, total);
}

// ---------- workspace layout (bytes) ----------
static const size_t OFF_SH1O  = 0;                               // E*3*4
static const size_t OFF_BASIS = OFF_SH1O  + (size_t)NEDGE*3*4;   // E*32*2
static const size_t OFF_H     = OFF_BASIS + (size_t)NEDGE*32*2;  // E*64*2
static const size_t OFF_W     = OFF_H     + (size_t)NEDGE*64*2;  // E*512*2
static const size_t OFF_WB1   = OFF_W     + (size_t)NEDGE*512*2; // 32*64*2 (swizzled)
static const size_t OFF_WB2   = OFF_WB1   + 32*64*2;             // 64*512*2 (swizzled)
static const size_t OFF_FEAT  = OFF_WB2   + 64*512*2;            // 8192*448*4
static const size_t OFF_SC    = OFF_FEAT  + (size_t)NNODE*448*4; // 8192*288*4
static const size_t OFF_AGG   = OFF_SC    + (size_t)NNODE*288*4; // 8192*1152*4
static const size_t OFF_Y     = OFF_AGG   + (size_t)NNODE*1152*4;// 8192*320*4
static const size_t OFF_GATED = OFF_Y     + (size_t)NNODE*320*4; // 8192*256*4
static const size_t OFF_GSUM  = OFF_GATED + (size_t)NNODE*256*4; // 1024*256*4
static const size_t OFF_CNT   = OFF_GSUM  + (size_t)NGRP*256*4;  // 1024*4
static const size_t OFF_FIN   = OFF_CNT   + (size_t)NGRP*4;      // 8192*512*4

extern "C" void kernel_launch(void* const* d_in, const int* in_sizes, int n_in,
                              void* d_out, int out_size, void* d_ws, size_t ws_size,
                              hipStream_t stream) {
  (void)in_sizes; (void)n_in; (void)out_size; (void)ws_size;
  const float CS = 0.3826834323650898f;   // sin(pi/8)
  const float CX = 0.9238795325112867f;   // cos(pi/8)

  const float* x      = (const float*)d_in[0];
  const float* pos    = (const float*)d_in[1];
  const float* eshift = (const float*)d_in[2];
  const float* lat    = (const float*)d_in[3];
  auto P = [&](int i) { return (const float*)d_in[i]; };
  const int* eidx  = (const int*)d_in[32];
  const int* batch = (const int*)d_in[33];
  const int* gidx  = (const int*)d_in[34];
  const int* esrc  = eidx;
  const int* edst  = eidx + NEDGE;

  char* ws = (char*)d_ws;
  float*          SH1O  = (float*)(ws + OFF_SH1O);
  unsigned short* BASIS = (unsigned short*)(ws + OFF_BASIS);
  unsigned short* HBUF  = (unsigned short*)(ws + OFF_H);
  unsigned short* WEDGE = (unsigned short*)(ws + OFF_W);
  unsigned short* WB1   = (unsigned short*)(ws + OFF_WB1);
  unsigned short* WB2   = (unsigned short*)(ws + OFF_WB2);
  float* FEAT  = (float*)(ws + OFF_FEAT);
  float* SC    = (float*)(ws + OFF_SC);
  float* AGG   = (float*)(ws + OFF_AGG);
  float* Y     = (float*)(ws + OFF_Y);
  float* GATED = (float*)(ws + OFF_GATED);
  float* GSUM  = (float*)(ws + OFF_GSUM);
  float* CNT   = (float*)(ws + OFF_CNT);
  float* FIN   = (float*)(ws + OFF_FIN);

  // group counts + edge embedding (shared by all convs)
  hipMemsetAsync(CNT, 0, (size_t)NGRP * 4, stream);
  count_kernel<<<NNODE / 256, 256, 0, stream>>>(gidx, CNT, NNODE);
  edge_embed_kernel<<<NEDGE / 256, 256, 0, stream>>>(pos, eshift, lat, batch, esrc, edst,
                                                     SH1O, BASIS, NEDGE);

  // ================= conv1 =================
  launch_packb(stream, P(11), WB1, 16, 64,  32, 0.25f);
  launch_packb(stream, P(12), WB2, 64, 160, 64, 0.125f);
  gemm_bf16_wmma_kernel<<<dim3(NEDGE/128, 4),  256, 0, stream>>>(BASIS, 32, WB1, HBUF,  64,  32, 1, 1);
  gemm_bf16_wmma_kernel<<<dim3(NEDGE/128, 10), 256, 0, stream>>>(HBUF,  64, WB2, WEDGE, 160, 64, 0, 1);
  // lin1 (feat layout: 0e@0 32d1, 1o@32 32d3; stride 128) ; input = x directly
  launch_lin(stream, x, 128, 0,  P(6), 32, 32, 1, FEAT, 128, 0,  nullptr, 0, 0, 0, 0);
  launch_lin(stream, x, 128, 32, P(7), 32, 32, 3, FEAT, 128, 32, nullptr, 0, 0, 0, 0);
  // sc (0e@0 96d1, 1o@96 32d3; stride 192)
  launch_lin(stream, x, 128, 0,  P(4), 96, 32, 1, SC, 192, 0,  nullptr, 0, 0, 0, 0);
  launch_lin(stream, x, 128, 32, P(5), 32, 32, 3, SC, 192, 96, nullptr, 0, 0, 0, 0);
  // TP + scatter (agg: 0e@0 64d1, 1o@64 64d3, 1e@256 32d3; stride 352)
  hipMemsetAsync(AGG, 0, (size_t)NNODE * 352 * 4, stream);
  {
    TPDesc d = {5, 128, 352, 160, 160, {
      {0,0,0, 32, 0,  0,   0},
      {0,1,1, 32, 0,  32,  64},
      {1,0,1, 32, 32, 64,  160},
      {1,1,0, 32, 32, 96,  32},
      {1,1,1, 32, 32, 128, 256} }};
    int tot = NEDGE * 160;
    tp_scatter_kernel<<<(tot+255)/256, 256, 0, stream>>>(esrc, edst, SH1O, WEDGE, FEAT, AGG, d, tot);
  }
  // lin2 + sc combine (Y: 0e@0 96d1, 1o@96 32d3, 1e@192 32d3; stride 288)
  launch_lin(stream, AGG, 352, 0,   P(8),  96, 64, 1, Y, 288, 0,   SC, 192, 0,  CS, CX);
  launch_lin(stream, AGG, 352, 64,  P(9),  32, 64, 3, Y, 288, 96,  SC, 192, 96, CS, CX);
  launch_lin(stream, AGG, 352, 256, P(10), 32, 32, 3, Y, 288, 192, nullptr, 0, 0, 0, 0);
  // gate1 + mix -> FIN (0e@0 64d1, 1o@64 64d3, 1e@256 64d3; stride 448)
  gate1_kernel<<<(NNODE*32+255)/256, 256, 0, stream>>>(Y, GATED, NNODE*32);
  hipMemsetAsync(GSUM, 0, (size_t)NGRP * 224 * 4, stream);
  gsum_scatter_kernel<<<(NNODE*224+255)/256, 256, 0, stream>>>(GATED, gidx, GSUM, 224, NNODE*224);
  launch_mix(stream, GATED, 224, 0,   GSUM, CNT, gidx, FIN, 448, 0,   32, 1);
  launch_mix(stream, GATED, 224, 32,  GSUM, CNT, gidx, FIN, 448, 64,  32, 3);
  launch_mix(stream, GATED, 224, 128, GSUM, CNT, gidx, FIN, 448, 256, 32, 3);

  // ================= conv2 =================
  launch_packb(stream, P(23), WB1, 16, 64,  32, 0.25f);
  launch_packb(stream, P(24), WB2, 64, 512, 64, 0.125f);
  gemm_bf16_wmma_kernel<<<dim3(NEDGE/128, 4),  256, 0, stream>>>(BASIS, 32, WB1, HBUF,  64,  32, 1, 1);
  gemm_bf16_wmma_kernel<<<dim3(NEDGE/128, 32), 256, 0, stream>>>(HBUF,  64, WB2, WEDGE, 512, 64, 0, 1);
  // lin1 (FIN stride 448: 0e@0, 1o@64, 1e@256) -> FEAT (same layout, 64ch each)
  launch_lin(stream, FIN, 448, 0,   P(16), 64, 64, 1, FEAT, 448, 0,   nullptr, 0, 0, 0, 0);
  launch_lin(stream, FIN, 448, 64,  P(17), 64, 64, 3, FEAT, 448, 64,  nullptr, 0, 0, 0, 0);
  launch_lin(stream, FIN, 448, 256, P(18), 64, 64, 3, FEAT, 448, 256, nullptr, 0, 0, 0, 0);
  // sc (0e@0 96d1, 1o@96 32d3, 1e@192 32d3; stride 288)
  launch_lin(stream, FIN, 448, 0,   P(13), 96, 64, 1, SC, 288, 0,   nullptr, 0, 0, 0, 0);
  launch_lin(stream, FIN, 448, 64,  P(14), 32, 64, 3, SC, 288, 96,  nullptr, 0, 0, 0, 0);
  launch_lin(stream, FIN, 448, 256, P(15), 32, 64, 3, SC, 288, 192, nullptr, 0, 0, 0, 0);
  // TP (agg: 0e@0 128d1, 1o@128 192d3, 1e@704 128d3, 0o@1088 64d1; stride 1152)
  hipMemsetAsync(AGG, 0, (size_t)NNODE * 1152 * 4, stream);
  {
    TPDesc d = {8, 448, 1152, 512, 512, {
      {0,0,0, 64, 0,   0,   0},
      {0,1,1, 64, 0,   64,  128},
      {1,0,1, 64, 64,  128, 320},
      {1,1,0, 64, 64,  192, 64},
      {1,1,1, 64, 64,  256, 704},
      {1,0,1, 64, 256, 320, 896},
      {1,1,0, 64, 256, 384, 1088},
      {1,1,1, 64, 256, 448, 512} }};
    int tot = NEDGE * 512;
    tp_scatter_kernel<<<(tot+255)/256, 256, 0, stream>>>(esrc, edst, SH1O, WEDGE, FEAT, AGG, d, tot);
  }
  // lin2 (Y: 0o@0 32d1, 0e@32 96d1, 1o@128 32d3, 1e@224 32d3; stride 320)
  launch_lin(stream, AGG, 1152, 1088, P(19), 32, 64,  1, Y, 320, 0,   nullptr, 0, 0, 0, 0);
  launch_lin(stream, AGG, 1152, 0,    P(20), 96, 128, 1, Y, 320, 32,  SC, 288, 0,  CS, CX);
  launch_lin(stream, AGG, 1152, 128,  P(21), 32, 192, 3, Y, 320, 128, SC, 288, 96, CS, CX);
  launch_lin(stream, AGG, 1152, 704,  P(22), 32, 128, 3, Y, 320, 224, SC, 288, 192, CS, CX);
  // gate2 + mix -> FIN (0o@0 64d1, 0e@64 64d1, 1o@128 64d3, 1e@320 64d3; stride 512)
  gate2_kernel<<<(NNODE*32+255)/256, 256, 0, stream>>>(Y, GATED, NNODE*32);
  hipMemsetAsync(GSUM, 0, (size_t)NGRP * 256 * 4, stream);
  gsum_scatter_kernel<<<(NNODE*256+255)/256, 256, 0, stream>>>(GATED, gidx, GSUM, 256, NNODE*256);
  launch_mix(stream, GATED, 256, 0,   GSUM, CNT, gidx, FIN, 512, 0,   32, 1);
  launch_mix(stream, GATED, 256, 32,  GSUM, CNT, gidx, FIN, 512, 64,  32, 1);
  launch_mix(stream, GATED, 256, 64,  GSUM, CNT, gidx, FIN, 512, 128, 32, 3);
  launch_mix(stream, GATED, 256, 160, GSUM, CNT, gidx, FIN, 512, 320, 32, 3);

  // ================= convf =================
  launch_packb(stream, P(30), WB1, 16, 64,  32, 0.25f);
  launch_packb(stream, P(31), WB2, 64, 192, 64, 0.125f);
  gemm_bf16_wmma_kernel<<<dim3(NEDGE/128, 4),  256, 0, stream>>>(BASIS, 32, WB1, HBUF,  64,  32, 1, 1);
  gemm_bf16_wmma_kernel<<<dim3(NEDGE/128, 12), 256, 0, stream>>>(HBUF,  64, WB2, WEDGE, 192, 64, 0, 1);
  // lin1 (FIN stride 512: 0e@64, 1o@128, 1e@320) -> FEAT (0e@0, 1o@64, 1e@256; stride 448)
  launch_lin(stream, FIN, 512, 64,  P(26), 64, 64, 1, FEAT, 448, 0,   nullptr, 0, 0, 0, 0);
  launch_lin(stream, FIN, 512, 128, P(27), 64, 64, 3, FEAT, 448, 64,  nullptr, 0, 0, 0, 0);
  launch_lin(stream, FIN, 512, 320, P(28), 64, 64, 3, FEAT, 448, 256, nullptr, 0, 0, 0, 0);
  // sc 1o: (1,64) -> SC as 8192 x 3
  launch_lin(stream, FIN, 512, 128, P(25), 1, 64, 3, SC, 3, 0, nullptr, 0, 0, 0, 0);
  // TP (agg: 1o 192d3; stride 576)
  hipMemsetAsync(AGG, 0, (size_t)NNODE * 576 * 4, stream);
  {
    TPDesc d = {3, 448, 576, 192, 192, {
      {0,1,1, 64, 0,   0,   0},
      {1,0,1, 64, 64,  64,  192},
      {1,1,1, 64, 256, 128, 384} }};
    int tot = NEDGE * 192;
    tp_scatter_kernel<<<(tot+255)/256, 256, 0, stream>>>(esrc, edst, SH1O, WEDGE, FEAT, AGG, d, tot);
  }
  // lin2 1o (192 -> 1) + sc combine -> d_out (8192 x 3)
  launch_lin(stream, AGG, 576, 0, P(29), 1, 192, 3, (float*)d_out, 3, 0, SC, 3, 0, CS, CX);
}
